// GNNCollisionPredictor_46823733461641
// MI455X (gfx1250) — matrix operand order repaired and checked by reading, not compiled, not run on hardware
//
#include <hip/hip_runtime.h>
#include <hip/hip_bf16.h>
#include <math.h>

#define NN 2048
#define PAIRS ((long)NN * (NN - 1) / 2)   // 2096128
#define NTILES (PAIRS / 16)               // 131008

typedef __attribute__((ext_vector_type(16))) __bf16 v16bf;
typedef __attribute__((ext_vector_type(2)))  __bf16 v2bf;
typedef __attribute__((ext_vector_type(8)))  float  v8f;

// packed float2 -> bf16x2 conversion: prefer native v_cvt_pk_bf16_f32
#if __has_builtin(__builtin_amdgcn_cvt_pk_bf16_f32)
static __device__ __forceinline__ v2bf pk2bf(float a, float b) {
  return __builtin_amdgcn_cvt_pk_bf16_f32(a, b);
}
#else
static __device__ __forceinline__ v2bf pk2bf(float a, float b) {
  v2bf r;
  r[0] = (__bf16)a;   // backend cvt (RNE); falls back to libcall-free lowering
  r[1] = (__bf16)b;
  return r;
}
#endif

// scalar float -> bf16 (RNE bit trick; used only in the one-time LDS preload)
static __device__ __forceinline__ __bf16 f2bf(float f) {
  unsigned u = __builtin_bit_cast(unsigned, f);
  unsigned r = u + 0x7FFFu + ((u >> 16) & 1u);
  unsigned short h = (unsigned short)(r >> 16);
  return __builtin_bit_cast(__bf16, h);
}

// pairs preceding row i in the upper-triangular (i<j) enumeration
static __device__ __forceinline__ long rowstart(int i) {
  return ((long)i * (2 * NN - 1 - i)) >> 1;
}

// s[n][h] = sum_k adj[n][k] * x[k][h]   (H = 4 or 64); one thread per output
__global__ __launch_bounds__(256) void adjmul_kernel(const float* __restrict__ adj,
                                                     const float* __restrict__ x,
                                                     float* __restrict__ s, int H) {
  int t = blockIdx.x * 256 + threadIdx.x;
  int n = t / H, h = t % H;
  const float* arow = adj + (long)n * NN;
  float acc = 0.f;
#pragma unroll 4
  for (int k = 0; k < NN; ++k)
    acc = fmaf(arow[k], x[k * H + h], acc);
  s[t] = acc;
}

// y[n][ho] = relu(b[ho] + sum_hi s[n][hi] * W[hi][ho]);  Hout = 64
__global__ __launch_bounds__(256) void linear_relu_kernel(const float* __restrict__ s,
                                                          const float* __restrict__ W,
                                                          const float* __restrict__ b,
                                                          float* __restrict__ y, int Hin) {
  int t = blockIdx.x * 256 + threadIdx.x;
  int n = t >> 6, ho = t & 63;
  const float* srow = s + (long)n * Hin;
  float acc = b[ho];
#pragma unroll 4
  for (int hi = 0; hi < Hin; ++hi)
    acc = fmaf(srow[hi], W[hi * 64 + ho], acc);
  y[t] = fmaxf(acc, 0.f);
}

// Apre[n][c] = be1[c] + sum_hi x3[n][hi]*We1[hi][c]
// Bpre[n][c] =          sum_hi x3[n][hi]*We1[64+hi][c]
__global__ __launch_bounds__(256) void edge_pre_kernel(const float* __restrict__ x3,
                                                       const float* __restrict__ We1,
                                                       const float* __restrict__ be1,
                                                       float* __restrict__ Apre,
                                                       float* __restrict__ Bpre) {
  int t = blockIdx.x * 256 + threadIdx.x;
  int n = t >> 7, c = t & 127;
  const float* xr = x3 + (long)n * 64;
  float a = be1[c], bb = 0.f;
#pragma unroll 4
  for (int hi = 0; hi < 64; ++hi) {
    float xv = xr[hi];
    a  = fmaf(xv, We1[hi * 128 + c], a);
    bb = fmaf(xv, We1[(64 + hi) * 128 + c], bb);
  }
  Apre[t] = a;
  Bpre[t] = bb;
}

// Per-wave 16-pair tile: H1(16x128,bf16) built in A-fragment layout,
// H2 = relu(H1 @ We2 + be2) via 16x v_wmma_f32_16x16x32_bf16,
// out = sigmoid(H2 @ We3 + be3) via shfl reduction.
__global__ __launch_bounds__(256) void pair_mlp_kernel(const float* __restrict__ Apre,
                                                       const float* __restrict__ Bpre,
                                                       const float* __restrict__ We2,
                                                       const float* __restrict__ be2,
                                                       const float* __restrict__ We3,
                                                       const float* __restrict__ be3,
                                                       float* __restrict__ out) {
  __shared__ __align__(32) __bf16 we2s[4 * 4 * 32 * 16]; // 8192 bf16 = 16KB, B-fragment order
  __shared__ float we3s[64];
  __shared__ float be2s[64];
  __shared__ float be3s;

  int tid = threadIdx.x;
  // Swizzle We2 (128x64 f32) into per-fragment contiguous bf16:
  // idx = ((kk*4+nn)*32 + lane)*16 + e  <->  We2[(kk*32 + 8*(lane>=16) + (e&7) + 16*(e>=8))*64 + nn*16 + (lane&15)]
  for (int idx = tid; idx < 8192; idx += 256) {
    int e  = idx & 15;
    int ln = (idx >> 4) & 31;
    int nn = (idx >> 9) & 3;
    int kk = (idx >> 11) & 3;
    int K  = kk * 32 + ((ln < 16) ? 0 : 8) + (e & 7) + ((e & 8) << 1);
    int Nc = nn * 16 + (ln & 15);
    we2s[idx] = f2bf(We2[K * 64 + Nc]);
  }
  if (tid < 64) { we3s[tid] = We3[tid]; be2s[tid] = be2[tid]; }
  if (tid == 0) be3s = be3[0];
  __syncthreads();

  const int lane   = tid & 31;
  const int wave   = tid >> 5;
  const int m      = lane & 15;      // tile row this lane builds (A-frag) / N-lane (C)
  const int half   = lane >> 4;      // 0: rows 0-7 of C, K-base 0; 1: rows 8-15, K-base 8
  const int base   = half * 8;       // A-fragment K base within a 32-wide K-step
  const int gwave  = blockIdx.x * 8 + wave;
  const int nwaves = gridDim.x * 8;

  const float bv0 = be2s[0 + m], bv1 = be2s[16 + m], bv2 = be2s[32 + m], bv3 = be2s[48 + m];
  const float w30 = we3s[0 + m], w31 = we3s[16 + m], w32 = we3s[32 + m], w33 = we3s[48 + m];

  for (long tile = gwave; tile < NTILES; tile += nwaves) {
    // invert triangular pair index p -> (i, j), per lane (row m of the tile)
    long p = tile * 16 + m;
    double q = 2.0 * (double)NN - 1.0;
    int i = (int)((q - sqrt(q * q - 8.0 * (double)p)) * 0.5);
    if (i < 0) i = 0;
    if (i > NN - 2) i = NN - 2;
    while (rowstart(i + 1) <= p) ++i;
    while (rowstart(i) > p) --i;
    int j = i + 1 + (int)(p - rowstart(i));

    const float* Ai = Apre + (long)i * 128;
    const float* Bj = Bpre + (long)j * 128;

    v8f c0, c1, c2, c3;
#pragma unroll
    for (int r = 0; r < 8; ++r) { c0[r] = bv0; c1[r] = bv1; c2[r] = bv2; c3[r] = bv3; }

#pragma unroll
    for (int kk = 0; kk < 4; ++kk) {
      const float4* a4 = (const float4*)(Ai + kk * 32 + base);
      const float4* b4 = (const float4*)(Bj + kk * 32 + base);
      float4 ta0 = a4[0], ta1 = a4[1], ta2 = a4[4], ta3 = a4[5];
      float4 tb0 = b4[0], tb1 = b4[1], tb2 = b4[4], tb3 = b4[5];
      // H1 = relu(Apre[i] + Bpre[j]) ; be1 already folded into Apre
      float h0  = fmaxf(ta0.x + tb0.x, 0.f), h1  = fmaxf(ta0.y + tb0.y, 0.f);
      float h2  = fmaxf(ta0.z + tb0.z, 0.f), h3  = fmaxf(ta0.w + tb0.w, 0.f);
      float h4  = fmaxf(ta1.x + tb1.x, 0.f), h5  = fmaxf(ta1.y + tb1.y, 0.f);
      float h6  = fmaxf(ta1.z + tb1.z, 0.f), h7  = fmaxf(ta1.w + tb1.w, 0.f);
      float h8  = fmaxf(ta2.x + tb2.x, 0.f), h9  = fmaxf(ta2.y + tb2.y, 0.f);
      float h10 = fmaxf(ta2.z + tb2.z, 0.f), h11 = fmaxf(ta2.w + tb2.w, 0.f);
      float h12 = fmaxf(ta3.x + tb3.x, 0.f), h13 = fmaxf(ta3.y + tb3.y, 0.f);
      float h14 = fmaxf(ta3.z + tb3.z, 0.f), h15 = fmaxf(ta3.w + tb3.w, 0.f);

      union { v16bf v; v2bf p[8]; } au;
      au.p[0] = pk2bf(h0,  h1);
      au.p[1] = pk2bf(h2,  h3);
      au.p[2] = pk2bf(h4,  h5);
      au.p[3] = pk2bf(h6,  h7);
      au.p[4] = pk2bf(h8,  h9);
      au.p[5] = pk2bf(h10, h11);
      au.p[6] = pk2bf(h12, h13);
      au.p[7] = pk2bf(h14, h15);
      v16bf af = au.v;

      v16bf bf0 = *(const v16bf*)(&we2s[((kk * 4 + 0) * 32 + lane) * 16]);
      v16bf bf1 = *(const v16bf*)(&we2s[((kk * 4 + 1) * 32 + lane) * 16]);
      v16bf bf2 = *(const v16bf*)(&we2s[((kk * 4 + 2) * 32 + lane) * 16]);
      v16bf bf3 = *(const v16bf*)(&we2s[((kk * 4 + 3) * 32 + lane) * 16]);
      c0 = __builtin_amdgcn_wmma_f32_16x16x32_bf16(false, af, false, bf0, (short)0, c0, false, false);
      c1 = __builtin_amdgcn_wmma_f32_16x16x32_bf16(false, af, false, bf1, (short)0, c1, false, false);
      c2 = __builtin_amdgcn_wmma_f32_16x16x32_bf16(false, af, false, bf2, (short)0, c2, false, false);
      c3 = __builtin_amdgcn_wmma_f32_16x16x32_bf16(false, af, false, bf3, (short)0, c3, false, false);
    }

    // Layer 3: out[M] = sigmoid(sum_n relu(H2[M][n]) * We3[n] + be3)
    // lane holds C rows M = r + 8*half, columns n = nn*16 + m
    float sacc[8];
#pragma unroll
    for (int r = 0; r < 8; ++r) {
      float h0 = fmaxf(c0[r], 0.f);
      float h1 = fmaxf(c1[r], 0.f);
      float h2 = fmaxf(c2[r], 0.f);
      float h3 = fmaxf(c3[r], 0.f);
      sacc[r] = h0 * w30 + h1 * w31 + h2 * w32 + h3 * w33;
    }
#pragma unroll
    for (int off = 1; off < 16; off <<= 1) {
#pragma unroll
      for (int r = 0; r < 8; ++r)
        sacc[r] += __shfl_xor(sacc[r], off, 32);
    }
    if (m == 0) {
      long pbase = tile * 16 + half * 8;
#pragma unroll
      for (int r = 0; r < 8; ++r) {
        float v = sacc[r] + be3s;
        out[pbase + r] = 1.0f / (1.0f + __expf(-v));
      }
    }
  }
}

extern "C" void kernel_launch(void* const* d_in, const int* in_sizes, int n_in,
                              void* d_out, int out_size, void* d_ws, size_t ws_size,
                              hipStream_t stream) {
  const float* nodef = (const float*)d_in[0];
  const float* adj   = (const float*)d_in[1];
  const float* W0    = (const float*)d_in[2];
  const float* b0    = (const float*)d_in[3];
  const float* W1    = (const float*)d_in[4];
  const float* b1    = (const float*)d_in[5];
  const float* W2    = (const float*)d_in[6];
  const float* b2    = (const float*)d_in[7];
  const float* We1   = (const float*)d_in[8];
  const float* be1   = (const float*)d_in[9];
  const float* We2   = (const float*)d_in[10];
  const float* be2   = (const float*)d_in[11];
  const float* We3   = (const float*)d_in[12];
  const float* be3   = (const float*)d_in[13];
  float* out = (float*)d_out;

  float* ws   = (float*)d_ws;
  float* sbuf = ws;                 // 2048*64
  float* x1   = ws + 131072;        // 2048*64 (also reused for x3)
  float* x2   = ws + 262144;        // 2048*64
  float* Apre = ws + 393216;        // 2048*128
  float* Bpre = ws + 655360;        // 2048*128
  float* x3   = x1;

  // GCN layer 0 (D=4 -> H=64)
  adjmul_kernel<<<NN * 4 / 256, 256, 0, stream>>>(adj, nodef, sbuf, 4);
  linear_relu_kernel<<<NN * 64 / 256, 256, 0, stream>>>(sbuf, W0, b0, x1, 4);
  // GCN layer 1
  adjmul_kernel<<<NN * 64 / 256, 256, 0, stream>>>(adj, x1, sbuf, 64);
  linear_relu_kernel<<<NN * 64 / 256, 256, 0, stream>>>(sbuf, W1, b1, x2, 64);
  // GCN layer 2
  adjmul_kernel<<<NN * 64 / 256, 256, 0, stream>>>(adj, x2, sbuf, 64);
  linear_relu_kernel<<<NN * 64 / 256, 256, 0, stream>>>(sbuf, W2, b2, x3, 64);
  // Edge predictor layer-1 factorization
  edge_pre_kernel<<<NN * 128 / 256, 256, 0, stream>>>(x3, We1, be1, Apre, Bpre);
  // All-pairs MLP via WMMA
  pair_mlp_kernel<<<512, 256, 0, stream>>>(Apre, Bpre, We2, be2, We3, be3, out);
}
